// MultiHeadAttention_14456859918347
// MI455X (gfx1250) — matrix operand order repaired and checked by reading, not compiled, not run on hardware
//
#include <hip/hip_runtime.h>
#include <hip/hip_bf16.h>

// MI455X (gfx1250) multi-head attention, bf16 WMMA pipeline.
// Compute-bound (~103 GFLOP vs ~100MB HBM traffic @ 23.3 TB/s), so all GEMM-like
// work goes through v_wmma_f32_16x16x32_bf16. Intermediates (~48 MiB) fit in L2.
// Flash-attention softmax: row-max via DPP butterflies (no LDS), row-sum via a
// WMMA against an all-ones fragment (layout-matched to the C fragment rows).

typedef __bf16 bf16;
typedef __attribute__((ext_vector_type(16))) __bf16 v16bf;
typedef __attribute__((ext_vector_type(8)))  __bf16 v8bf;
typedef __attribute__((ext_vector_type(4)))  __bf16 v4bf;
typedef __attribute__((ext_vector_type(8)))  float  v8f;

#define T_SEQ   4096
#define C_DIM   1024
#define H_HEADS 16
#define D_HEAD  64

union ABu { v16bf v; v8bf h[2]; };

// A-fragment (16x32 bf16, MxK) from row-major memory, per CDNA5 ISA:
// lane L: row m = L&15; lanes 0-15 hold K 0..7 & 16..23, lanes 16-31 K 8..15 & 24..31.
__device__ __forceinline__ v16bf load_a_frag(const bf16* base, int ld, int lane) {
  const int m  = lane & 15;
  const int kb = (lane >> 4) << 3;
  ABu u;
  u.h[0] = *(const v8bf*)(base + (size_t)m * ld + kb);
  u.h[1] = *(const v8bf*)(base + (size_t)m * ld + kb + 16);
  return u.v;
}

// B-fragment (32x16 bf16, KxN): lane L holds B[kb..kb+7][n] and B[kb+16..kb+23][n], n = L&15, kb = 8*(L>>4).
// `base + n*ld + k` must address contiguous bf16 along K (true for W[out][in] row-major,
// K rows of Kb, and rows of transposed Vt).
__device__ __forceinline__ v16bf load_b_frag(const bf16* base, int ld, int lane) {
  const int n  = lane & 15;
  const int kb = (lane >> 4) << 3;
  ABu u;
  u.h[0] = *(const v8bf*)(base + (size_t)n * ld + kb);
  u.h[1] = *(const v8bf*)(base + (size_t)n * ld + kb + 16);
  return u.v;
}

__device__ __forceinline__ v8f wmma_bf16(v16bf a, v16bf b, v8f c) {
  return __builtin_amdgcn_wmma_f32_16x16x32_bf16(false, a, false, b, (short)0, c,
                                                 false, false);
}

// 16-lane max reduction with DPP16 (stays within each 16-lane half of wave32):
// quad swap1 (0xB1), quad swap2 (0x4E), row_half_mirror (0x141), row_mirror (0x140).
template <int CTRL>
__device__ __forceinline__ float dpp_maxstep(float x) {
  int i = __builtin_bit_cast(int, x);
  int p = __builtin_amdgcn_update_dpp(i, i, CTRL, 0xf, 0xf, true);
  return fmaxf(x, __builtin_bit_cast(float, p));
}
__device__ __forceinline__ float rowmax16(float x) {
  x = dpp_maxstep<0xB1>(x);   // lane ^ 1
  x = dpp_maxstep<0x4E>(x);   // lane ^ 2
  x = dpp_maxstep<0x141>(x);  // half-row mirror (crosses quads)
  x = dpp_maxstep<0x140>(x);  // row mirror (crosses 8-groups)
  return x;
}

// ---------------------------------------------------------------- cast f32->bf16
__global__ __launch_bounds__(256) void cast_f32_bf16(const float* __restrict__ in,
                                                     bf16* __restrict__ out, int n4) {
  int i = blockIdx.x * 256 + threadIdx.x;
  if (i >= n4) return;
  float4 f = ((const float4*)in)[i];
  v4bf o;
  o.x = (bf16)f.x; o.y = (bf16)f.y; o.z = (bf16)f.z; o.w = (bf16)f.w;
  ((v4bf*)out)[i] = o;
}

// ---------------------------------------------------------------- GEMM: y = (A @ W^T + b) * scale
// A: [4096,1024] bf16 row-major. W: [1024,1024] bf16 row-major ([out][in], so
// y[m][n] = sum_k A[m][k]*W[n][k] -> B fragments are contiguous rows of W).
// MODE 0: bf16 row-major out [T,C]   (Q pre-scaled by 1/sqrt(D), K)
// MODE 1: bf16 transposed out [C,T]  (Vt: Vt[n][t], n = h*64+d)
// MODE 2: f32 row-major out [T,C]    (final projection)
// Block = 4 waves; each wave computes a 16x64 tile; grid (T/64, C/64).
template <int MODE>
__global__ __launch_bounds__(128) void gemm_xwT(const bf16* __restrict__ A,
                                                const bf16* __restrict__ W,
                                                const float* __restrict__ bias,
                                                void* __restrict__ outp,
                                                float scale) {
  const int lane = threadIdx.x & 31;
  const int wave = threadIdx.x >> 5;
  const int m0 = blockIdx.x * 64 + wave * 16;
  const int n0 = blockIdx.y * 64;

  v8f acc[4] = {};
  for (int k0 = 0; k0 < C_DIM; k0 += 32) {
    v16bf a = load_a_frag(A + (size_t)m0 * C_DIM + k0, C_DIM, lane);
#pragma unroll
    for (int nt = 0; nt < 4; ++nt) {
      v16bf b = load_b_frag(W + (size_t)(n0 + nt * 16) * C_DIM + k0, C_DIM, lane);
      acc[nt] = wmma_bf16(a, b, acc[nt]);
    }
  }

  const int nn = lane & 15;
  const int hl = lane >> 4;
#pragma unroll
  for (int nt = 0; nt < 4; ++nt) {
    const int n = n0 + nt * 16 + nn;
    const float bv = bias[n];
#pragma unroll
    for (int v = 0; v < 8; ++v) {
      const int row = m0 + v + 8 * hl;          // C-layout: row = vgpr + 8*(lane>=16)
      const float val = (acc[nt][v] + bv) * scale;
      if (MODE == 0)
        ((bf16*)outp)[(size_t)row * C_DIM + n] = (bf16)val;
      else if (MODE == 1)
        ((bf16*)outp)[(size_t)n * T_SEQ + row] = (bf16)val;
      else
        ((float*)outp)[(size_t)row * C_DIM + n] = val;
    }
  }
}

// ---------------------------------------------------------------- flash attention
// Grid (T/64, H). Block = 4 waves; wave w owns 16 query rows q0..q0+15 of head h.
// Streams keys in chunks of 32: S = Q@K^T via 4 WMMAs (Q pre-scaled by 1/sqrt(D)),
// online softmax: row-max via DPP, P written bf16 to LDS (C->A relayout),
// row-sum = WMMA(P, ones), O += P@V via 4 WMMAs against transposed Vt.
__global__ __launch_bounds__(128) void flash_attn(const bf16* __restrict__ Qb,
                                                  const bf16* __restrict__ Kb,
                                                  const bf16* __restrict__ Vt,
                                                  bf16* __restrict__ Ob) {
  __shared__ bf16 lp[4 * 16 * 32];  // 1KB per wave: P tile 16x32 bf16
  const int lane = threadIdx.x & 31;
  const int wave = threadIdx.x >> 5;
  const int q0 = blockIdx.x * 64 + wave * 16;
  const int h  = blockIdx.y;
  const int nn = lane & 15;
  const int hl = lane >> 4;
  bf16* lpw = lp + wave * 512;

  // Q tile 16x64 = two A fragments (K = d in [0,32) and [32,64)), loaded once.
  v16bf qa0 = load_a_frag(Qb + (size_t)q0 * C_DIM + h * 64,      C_DIM, lane);
  v16bf qa1 = load_a_frag(Qb + (size_t)q0 * C_DIM + h * 64 + 32, C_DIM, lane);

  // all-ones B fragment: WMMA(P, ones) -> every column n holds rowsum(m),
  // replicated across lanes in exactly the C-fragment row layout of l8[].
  v16bf ones;
#pragma unroll
  for (int i = 0; i < 16; ++i) ones[i] = (bf16)1.0f;

  float m8[8], l8[8], al[8];
  v8f o[4] = {};
#pragma unroll
  for (int v = 0; v < 8; ++v) { m8[v] = -1e30f; l8[v] = 0.f; }

  for (int j0 = 0; j0 < T_SEQ; j0 += 32) {
    // ---- S = Q @ K^T (already scaled), two 16x16 key tiles
    v8f s[2];
#pragma unroll
    for (int nt = 0; nt < 2; ++nt) {
      v16bf b0 = load_b_frag(Kb + (size_t)(j0 + nt * 16) * C_DIM + h * 64,      C_DIM, lane);
      v16bf b1 = load_b_frag(Kb + (size_t)(j0 + nt * 16) * C_DIM + h * 64 + 32, C_DIM, lane);
      v8f c = {};
      c = wmma_bf16(qa0, b0, c);
      c = wmma_bf16(qa1, b1, c);
      s[nt] = c;
    }

    // ---- online softmax; each lane-half owns rows {v + 8*hl}, col = nn
#pragma unroll
    for (int v = 0; v < 8; ++v) {
      const float rmax = rowmax16(fmaxf(s[0][v], s[1][v]));
      const float mn = fmaxf(m8[v], rmax);
      al[v] = __expf(m8[v] - mn);
      m8[v] = mn;
      const float p0 = __expf(s[0][v] - mn);
      const float p1 = __expf(s[1][v] - mn);
#pragma unroll
      for (int dt = 0; dt < 4; ++dt) o[dt][v] *= al[v];
      // write P (bf16) into this wave's LDS tile, row-major 16x32
      const int m = v + 8 * hl;
      lpw[m * 32 + nn]      = (bf16)p0;
      lpw[m * 32 + 16 + nn] = (bf16)p1;
    }

    // make cross-lane LDS stores visible before A-layout reload
    asm volatile("s_wait_dscnt 0" ::: "memory");

    ABu u;
    const int mm = lane & 15;
    const int kb = hl * 8;
    u.h[0] = *(const v8bf*)(lpw + mm * 32 + kb);
    u.h[1] = *(const v8bf*)(lpw + mm * 32 + kb + 16);
    v16bf pa = u.v;

    // ---- row sums of P in one WMMA (replaces 32 ds_bpermute butterflies)
    v8f zero = {};
    v8f rs = wmma_bf16(pa, ones, zero);
#pragma unroll
    for (int v = 0; v < 8; ++v) l8[v] = l8[v] * al[v] + rs[v];

    // ---- O += P @ V  (B fragments from transposed Vt: contiguous 32B/lane)
#pragma unroll
    for (int dt = 0; dt < 4; ++dt) {
      v16bf vb = load_b_frag(Vt + (size_t)(h * 64 + dt * 16) * T_SEQ + j0, T_SEQ, lane);
      o[dt] = wmma_bf16(pa, vb, o[dt]);
    }
  }

  // ---- normalize and store O as bf16 into [T, C] at column block h*64
#pragma unroll
  for (int v = 0; v < 8; ++v) {
    const float inv = 1.0f / l8[v];
    const int row = q0 + v + 8 * hl;
#pragma unroll
    for (int dt = 0; dt < 4; ++dt)
      Ob[(size_t)row * C_DIM + h * 64 + dt * 16 + nn] = (bf16)(o[dt][v] * inv);
  }
}

// ---------------------------------------------------------------- launch
extern "C" void kernel_launch(void* const* d_in, const int* in_sizes, int n_in,
                              void* d_out, int out_size, void* d_ws, size_t ws_size,
                              hipStream_t stream) {
  (void)in_sizes; (void)n_in; (void)out_size; (void)ws_size;
  const float* x  = (const float*)d_in[0];
  const float* Wq = (const float*)d_in[1];
  const float* bq = (const float*)d_in[2];
  const float* Wk = (const float*)d_in[3];
  const float* bk = (const float*)d_in[4];
  const float* Wv = (const float*)d_in[5];
  const float* bv = (const float*)d_in[6];
  const float* Wo = (const float*)d_in[7];
  const float* bo = (const float*)d_in[8];

  char* ws = (char*)d_ws;
  size_t off = 0;
  auto take = [&](size_t bytes) {
    char* p = ws + off;
    off = (off + bytes + 255) & ~(size_t)255;
    return p;
  };
  bf16* xb  = (bf16*)take((size_t)T_SEQ * C_DIM * 2);
  bf16* wqb = (bf16*)take((size_t)C_DIM * C_DIM * 2);
  bf16* wkb = (bf16*)take((size_t)C_DIM * C_DIM * 2);
  bf16* wvb = (bf16*)take((size_t)C_DIM * C_DIM * 2);
  bf16* wob = (bf16*)take((size_t)C_DIM * C_DIM * 2);
  bf16* Qb  = (bf16*)take((size_t)T_SEQ * C_DIM * 2);
  bf16* Kb  = (bf16*)take((size_t)T_SEQ * C_DIM * 2);
  bf16* Vt  = (bf16*)take((size_t)T_SEQ * C_DIM * 2);
  bf16* Ob  = (bf16*)take((size_t)T_SEQ * C_DIM * 2);

  const int nX = T_SEQ * C_DIM / 4, nW = C_DIM * C_DIM / 4;
  cast_f32_bf16<<<(nX + 255) / 256, 256, 0, stream>>>(x,  xb,  nX);
  cast_f32_bf16<<<(nW + 255) / 256, 256, 0, stream>>>(Wq, wqb, nW);
  cast_f32_bf16<<<(nW + 255) / 256, 256, 0, stream>>>(Wk, wkb, nW);
  cast_f32_bf16<<<(nW + 255) / 256, 256, 0, stream>>>(Wv, wvb, nW);
  cast_f32_bf16<<<(nW + 255) / 256, 256, 0, stream>>>(Wo, wob, nW);

  const float qscale = 0.125f;  // 1/sqrt(D_HEAD), folded into Q projection
  dim3 g(T_SEQ / 64, C_DIM / 64), b(128);
  gemm_xwT<0><<<g, b, 0, stream>>>(xb, wqb, bq, Qb, qscale);
  gemm_xwT<0><<<g, b, 0, stream>>>(xb, wkb, bk, Kb, 1.0f);
  gemm_xwT<1><<<g, b, 0, stream>>>(xb, wvb, bv, Vt, 1.0f);

  dim3 ga(T_SEQ / 64, H_HEADS);
  flash_attn<<<ga, b, 0, stream>>>(Qb, Kb, Vt, Ob);

  gemm_xwT<2><<<g, b, 0, stream>>>(Ob, wob, bo, d_out, 1.0f);
}
